// GenericEchoState_36240934044414
// MI455X (gfx1250) — matrix operand order repaired and checked by reading, not compiled, not run on hardware
//
#include <hip/hip_runtime.h>
#include <hip/hip_bf16.h>

// Echo-state generative rollout for MI455X (gfx1250).
//
// Weights converted once per launch to bf16 in workspace (~35.7 MB, resident
// in the 192MB L2), then one persistent kernel runs all n_steps with
// device-wide barriers; mat-vecs run on v_wmma_f32_16x16x32_bf16 with
// depth-1 software pipelining (loads for chunk c+1 in flight during the
// WMMA of chunk c).

#define N_RES 4096
#define N_OUT 128
#define N_CAT (N_RES + N_OUT) // 4224

typedef __attribute__((ext_vector_type(16))) __bf16 v16bf;
typedef __attribute__((ext_vector_type(8)))  float  v8f;

union BfPack { uint4 q[2]; v16bf v; };

struct MvRegs { uint4 a0, a1, b0, b1; };

__device__ __forceinline__ unsigned short f2bf(float f) {
  unsigned u = __builtin_bit_cast(unsigned, f);
  u += 0x7FFFu + ((u >> 16) & 1u); // round-to-nearest-even
  return (unsigned short)(u >> 16);
}

// Load one K=32 chunk: A = state slice (lanes 0/16 carry data after masking),
// B = 32 contiguous bf16 of weight row (rowBase + (lane&15)).
// A layout (16-bit A 16x32): lane<16 holds K{0..7,16..23}, lane>=16 K{8..15,24..31}
// B layout (16-bit B 32x16): lane n<16 -> K{0..7,16..23} of column n; lane n+16 -> K{8..15,24..31}
__device__ __forceinline__ MvRegs mv_load(const unsigned short* __restrict__ wrow, int boff,
                                          const unsigned short* __restrict__ svec, int aoff,
                                          int hi) {
  MvRegs r;
  r.a0 = *(const uint4*)(svec + aoff + hi * 8);
  r.a1 = *(const uint4*)(svec + aoff + 16 + hi * 8);
  r.b0 = *(const uint4*)(wrow + boff + hi * 8);
  r.b1 = *(const uint4*)(wrow + boff + 16 + hi * 8);
  return r;
}

// One v_wmma_f32_16x16x32_bf16: acc[0] (lanes 0..15) += dot(state, Wrow(lane)).
// Only A's M=0 row is live; other lanes are zeroed with ANDs so EXEC stays
// all-ones as WMMA requires.
__device__ __forceinline__ v8f mv_fma(MvRegs r, unsigned keep, v8f acc) {
  BfPack a, b;
  a.q[0] = r.a0; a.q[1] = r.a1;
  a.q[0].x &= keep; a.q[0].y &= keep; a.q[0].z &= keep; a.q[0].w &= keep;
  a.q[1].x &= keep; a.q[1].y &= keep; a.q[1].z &= keep; a.q[1].w &= keep;
  b.q[0] = r.b0; b.q[1] = r.b1;
  return __builtin_amdgcn_wmma_f32_16x16x32_bf16(false, a.v, false, b.v,
                                                 (short)0, acc, false, false);
}

// A-source for the concatenated [s_new ; y] readout input.
__device__ __forceinline__ const unsigned short* catsrc(const unsigned short* __restrict__ s,
                                                        const unsigned short* __restrict__ y,
                                                        int c) {
  return (c < 128) ? (s + c * 32) : (y + (c - 128) * 32);
}

// Sense-reversing device-wide barrier (bar[0]=arrive count, bar[1]=generation).
__device__ __forceinline__ void grid_barrier(unsigned* bar, unsigned nwg) {
  __syncthreads();
  __threadfence(); // publish this WG's global stores
  if (threadIdx.x == 0) {
    unsigned gen = __atomic_load_n(bar + 1, __ATOMIC_RELAXED);
    unsigned prev = __atomic_fetch_add(bar, 1u, __ATOMIC_ACQ_REL);
    if (prev == nwg - 1u) {
      __atomic_store_n(bar, 0u, __ATOMIC_RELAXED);
      __atomic_store_n(bar + 1, gen + 1u, __ATOMIC_RELEASE);
    } else {
      while (__atomic_load_n(bar + 1, __ATOMIC_ACQUIRE) == gen) {
        __builtin_amdgcn_s_sleep(1);
      }
    }
  }
  __threadfence();
  __syncthreads();
}

// ---- fp32 -> bf16 weight conversion (once per launch; trivial vs rollout) ----
__global__ void esn_convert(const float* __restrict__ wres_f,
                            const float* __restrict__ win_f,
                            const float* __restrict__ wout_f,
                            unsigned short* __restrict__ wres,
                            unsigned short* __restrict__ win,
                            unsigned short* __restrict__ wout) {
  size_t i = (size_t)blockIdx.x * blockDim.x + threadIdx.x;
  size_t stride = (size_t)gridDim.x * blockDim.x;
  for (size_t k = i; k < (size_t)N_RES * N_RES; k += stride) wres[k] = f2bf(wres_f[k]);
  for (size_t k = i; k < (size_t)N_RES * N_OUT; k += stride) win[k]  = f2bf(win_f[k]);
  for (size_t k = i; k < (size_t)N_OUT * N_CAT; k += stride) wout[k] = f2bf(wout_f[k]);
}

__global__ void esn_init(const float* __restrict__ state,
                         const float* __restrict__ initial,
                         unsigned short* __restrict__ s0,
                         unsigned short* __restrict__ y0,
                         unsigned* bar) {
  int i = threadIdx.x;
  for (int k = i; k < N_RES; k += 256) s0[k] = f2bf(state[k]);
  if (i < N_OUT) y0[i] = f2bf(initial[i]);
  if (i == 0) { bar[0] = 0u; bar[1] = 0u; }
}

// ---- Persistent rollout kernel: 64 WGs x 256 threads (8 waves) ----
// Phase A: each WG owns 64 reservoir rows (4 tiles of 16 rows); each tile is
//          computed by a wave pair splitting K=4096, reduced through LDS.
// Phase B: WGs 0..7 compute 16 readout rows each; 8 waves split the 132
//          K-chunks of W_out@[s_new;y], reduced through LDS.
// sbase holds BOTH s ping-pong buffers (sbase, sbase+N_RES); ybase likewise.
// Buffer selection is arithmetic so address-space inference keeps every
// access on the GLOBAL path (no flat_load).
__global__ void __launch_bounds__(256, 1)
esn_rollout(const unsigned short* __restrict__ wres,
            const unsigned short* __restrict__ win,
            const unsigned short* __restrict__ wout,
            const float* __restrict__ b_res,
            const float* __restrict__ b_out,
            unsigned short* sbase,
            unsigned short* ybase,
            unsigned* bar,
            float* __restrict__ out,
            const int* __restrict__ n_steps_ptr) {
  const int tid  = threadIdx.x;
  const int lane = tid & 31;
  const int wave = tid >> 5;          // 0..7
  const int hi   = lane >> 4;         // half-wave selector
  const unsigned keep = ((lane & 15) == 0) ? 0xFFFFFFFFu : 0u;
  const int nsteps = *n_steps_ptr;
  const unsigned nwg = gridDim.x;

  __shared__ float red[8][16];

  // Phase-A assignment
  const int tile = wave >> 1;         // 0..3
  const int half = wave & 1;          // K-split
  const int rowA = blockIdx.x * 64 + tile * 16 + (lane & 15);
  const unsigned short* wresRow = wres + (size_t)rowA * N_RES;
  const unsigned short* winRow  = win  + (size_t)rowA * N_OUT;
  // Phase-B assignment (meaningful for blockIdx.x < 8)
  const int rowB = (blockIdx.x & 7) * 16 + (lane & 15);
  const unsigned short* woutRow = wout + (size_t)rowB * N_CAT;

  for (int t = 0; t < nsteps; ++t) {
    const unsigned short* scur = sbase + (size_t)(t & 1) * N_RES;
    unsigned short*       snew = sbase + (size_t)((t + 1) & 1) * N_RES;
    const unsigned short* ycur = ybase + (size_t)(t & 1) * N_OUT;
    unsigned short*       ynew = ybase + (size_t)((t + 1) & 1) * N_OUT;

    // ---- Phase A: s_new = tanh(W_res @ s + W_in @ y + b_res) ----
    // 64 chunks of K=32 per wave (half of K=4096), depth-1 pipelined.
    v8f acc = {};
    {
      const int c0 = half * 64;
      MvRegs cur = mv_load(wresRow, c0 * 32, scur, c0 * 32, hi);
      #pragma unroll 4
      for (int c = c0; c < c0 + 63; ++c) {
        MvRegs nxt = mv_load(wresRow, (c + 1) * 32, scur, (c + 1) * 32, hi);
        acc = mv_fma(cur, keep, acc);
        cur = nxt;
      }
      acc = mv_fma(cur, keep, acc);
    }
    if (half == 0) { // W_in @ y (K=128): 4 chunks
      MvRegs cur = mv_load(winRow, 0, ycur, 0, hi);
      #pragma unroll
      for (int c = 0; c < 3; ++c) {
        MvRegs nxt = mv_load(winRow, (c + 1) * 32, ycur, (c + 1) * 32, hi);
        acc = mv_fma(cur, keep, acc);
        cur = nxt;
      }
      acc = mv_fma(cur, keep, acc);
    }
    if (lane < 16) red[wave][lane] = acc[0];
    __syncthreads();
    if (tid < 64) {
      int t2 = tid >> 4, n = tid & 15;
      int row = blockIdx.x * 64 + t2 * 16 + n;
      float z = red[2 * t2][n] + red[2 * t2 + 1][n] + b_res[row];
      snew[row] = f2bf(tanhf(z));
    }
    grid_barrier(bar, nwg);

    // ---- Phase B: pred = W_out @ [s_new ; y] + b_out ----
    if (blockIdx.x < 8) {
      v8f pacc = {};
      int c = wave; // 132 chunks total: 128 from s_new, 4 from y; stride 8
      MvRegs cur = mv_load(woutRow, c * 32, catsrc(snew, ycur, c), 0, hi);
      for (; c + 8 < 132; c += 8) {
        MvRegs nxt = mv_load(woutRow, (c + 8) * 32, catsrc(snew, ycur, c + 8), 0, hi);
        pacc = mv_fma(cur, keep, pacc);
        cur = nxt;
      }
      pacc = mv_fma(cur, keep, pacc);
      if (lane < 16) red[wave][lane] = pacc[0];
      __syncthreads();
      if (tid < 16) {
        int oc = (blockIdx.x << 4) + tid;
        float z = b_out[oc];
        #pragma unroll
        for (int w = 0; w < 8; ++w) z += red[w][tid];
        out[(size_t)t * N_OUT + oc] = z;   // fp32 prediction output
        ynew[oc] = f2bf(z);                // bf16 feedback for next step
      }
    }
    grid_barrier(bar, nwg);
  }
}

extern "C" void kernel_launch(void* const* d_in, const int* in_sizes, int n_in,
                              void* d_out, int out_size, void* d_ws, size_t ws_size,
                              hipStream_t stream) {
  const float* state   = (const float*)d_in[0];
  const float* initial = (const float*)d_in[1];
  const float* W_res   = (const float*)d_in[2];
  const float* W_in    = (const float*)d_in[3];
  const float* b_res   = (const float*)d_in[4];
  const float* W_out   = (const float*)d_in[5];
  const float* b_out   = (const float*)d_in[6];
  const int*   n_steps = (const int*)d_in[7];
  float* out = (float*)d_out;

  // Workspace layout (bf16 unless noted). Requires ~35.8 MB.
  unsigned short* ws      = (unsigned short*)d_ws;
  unsigned short* wres_bf = ws;
  unsigned short* win_bf  = wres_bf + (size_t)N_RES * N_RES;
  unsigned short* wout_bf = win_bf  + (size_t)N_RES * N_OUT;
  unsigned short* sbase   = wout_bf + (size_t)N_OUT * N_CAT; // 2 x N_RES (ping-pong)
  unsigned short* ybase   = sbase + 2 * N_RES;               // 2 x N_OUT (ping-pong)
  unsigned*       bar     = (unsigned*)(ybase + 2 * N_OUT);

  esn_convert<<<2048, 256, 0, stream>>>(W_res, W_in, W_out, wres_bf, win_bf, wout_bf);
  esn_init<<<1, 256, 0, stream>>>(state, initial, sbase, ybase, bar);
  esn_rollout<<<64, 256, 0, stream>>>(wres_bf, win_bf, wout_bf, b_res, b_out,
                                      sbase, ybase, bar, out, n_steps);
  (void)in_sizes; (void)n_in; (void)out_size; (void)ws_size;
}
